// ProbabilisticTransformerBlock_21165598835352
// MI455X (gfx1250) — compile-verified, matrix operand order checked by reading, not run on hardware
//
#include <hip/hip_runtime.h>
#include <hip/hip_bf16.h>

typedef __bf16 bf16;
typedef __attribute__((ext_vector_type(16))) __bf16 v16bf;
typedef __attribute__((ext_vector_type(8)))  __bf16 v8bf;
typedef __attribute__((ext_vector_type(8)))  float  v8f;
typedef unsigned int u32;
typedef __attribute__((ext_vector_type(4))) u32 u32x4;
typedef __attribute__((ext_vector_type(8))) u32 u32x8;

#define BSZ    2
#define LSEQ   2048
#define DMODEL 512
#define NHEAD  8
#define DK     64
#define NTOK   (BSZ * LSEQ)     // 4096
#define QKVN   (3 * DMODEL)     // 1536
#define DFF    (4 * DMODEL)     // 2048

// ---------------- WMMA helpers (CDNA5 16x16x32 bf16, f32 acc) ----------------

__device__ __forceinline__ v16bf ld_frag(const bf16* base, int hi) {
  // 16-bit A/B fragment layout: lane half `hi` holds K = hi*8..hi*8+7 and
  // K = 16+hi*8..16+hi*8+7 -> two contiguous 16B loads.
  v8bf lo = *(const v8bf*)(base + hi * 8);
  v8bf hh = *(const v8bf*)(base + 16 + hi * 8);
  return __builtin_shufflevector(lo, hh, 0, 1, 2, 3, 4, 5, 6, 7,
                                 8, 9, 10, 11, 12, 13, 14, 15);
}

__device__ __forceinline__ v8f wmma_bf16(v16bf a, v16bf b, v8f c) {
  return __builtin_amdgcn_wmma_f32_16x16x32_bf16(false, a, false, b,
                                                 (short)0, c, false, false);
}

// ---------------- Tensor Data Mover: 2D tile (128 rows x 64B) -> padded LDS --
// D# per CDNA5 ISA: group0 = count/lds/global/type,
// group1 = data_size + pad fields + tile/tensor dims + dim0 stride.
// Pads 16B after every 64B row -> matches As[128][40] (80B row stride).
__device__ __forceinline__ void tdm_load_a_tile(const bf16* gsrc, u32 lds_addr,
                                                int K) {
  unsigned long long ga = (unsigned long long)(uintptr_t)gsrc;
  u32 ga_lo = (u32)__builtin_amdgcn_readfirstlane((int)(u32)ga);
  u32 ga_hi = (u32)__builtin_amdgcn_readfirstlane((int)(u32)(ga >> 32));
  u32 lds   = (u32)__builtin_amdgcn_readfirstlane((int)lds_addr);
  u32 kk    = (u32)__builtin_amdgcn_readfirstlane((int)K);

  u32x4 g0;
  g0[0] = 1u;                                   // count=1 (user descriptor)
  g0[1] = lds;                                  // lds_addr (bytes)
  g0[2] = ga_lo;                                // global_addr[31:0]
  g0[3] = (ga_hi & 0x01FFFFFFu) | (2u << 30);   // global_addr[56:32] | type=2

  u32x8 g1;
  // data_size=1 (2B) | pad_enable | pad_interval=3 (16 DW) | pad_amount=3 (4 DW)
  g1[0] = (1u << 16) | (1u << 20) | (3u << 22) | (3u << 25);
  g1[1] = 0u;                                   // abar=0, tensor_dim0[15:0]=0
  g1[2] = 0x4000u;                              // tensor_dim0=2^30, td1 lo=0
  g1[3] = 0x4000u | (32u << 16);                // tensor_dim1=2^30, tile_dim0=32
  g1[4] = 128u;                                 // tile_dim1=128, tile_dim2=0
  g1[5] = kk;                                   // tensor_dim0_stride = K elems
  g1[6] = 0u;
  g1[7] = 0u;
  asm volatile("tensor_load_to_lds %0, %1" :: "s"(g0), "s"(g1) : "memory");
}

// ---------------- elementwise kernels ----------------

__global__ void cvt_bf16_kernel(const float* __restrict__ x,
                                bf16* __restrict__ y, int n) {
  int i = blockIdx.x * blockDim.x + threadIdx.x;
  if (i < n) y[i] = (bf16)x[i];
}

__global__ void softplus_bf16_kernel(const float* __restrict__ x,
                                     bf16* __restrict__ y, int n) {
  int i = blockIdx.x * blockDim.x + threadIdx.x;
  if (i < n) {
    float v = x[i];
    float sp = (v > 15.0f) ? v : log1pf(__expf(v));
    y[i] = (bf16)sp;
  }
}

// One block (256 threads) per row of D=512. LayerNorm -> bf16.
__global__ __launch_bounds__(256) void ln_bf16_kernel(
    const float* __restrict__ x, const float* __restrict__ g,
    const float* __restrict__ b, bf16* __restrict__ y) {
  __shared__ float red[256];
  int row = blockIdx.x, tid = threadIdx.x;
  const float* xr = x + (size_t)row * DMODEL;
  float a0 = xr[tid], a1 = xr[tid + 256];

  red[tid] = a0 + a1;
  __syncthreads();
  for (int o = 128; o > 0; o >>= 1) {
    if (tid < o) red[tid] += red[tid + o];
    __syncthreads();
  }
  float mean = red[0] * (1.0f / DMODEL);
  __syncthreads();

  float d0 = a0 - mean, d1 = a1 - mean;
  red[tid] = d0 * d0 + d1 * d1;
  __syncthreads();
  for (int o = 128; o > 0; o >>= 1) {
    if (tid < o) red[tid] += red[tid + o];
    __syncthreads();
  }
  float rstd = rsqrtf(red[0] * (1.0f / DMODEL) + 1e-5f);

  bf16* yr = y + (size_t)row * DMODEL;
  yr[tid]       = (bf16)(d0 * rstd * g[tid] + b[tid]);
  yr[tid + 256] = (bf16)(d1 * rstd * g[tid + 256] + b[tid + 256]);
}

// Per (token, head): qn = sum(Qm^2+Qs^2), kn = sum(Km^2+Ks^2) over dk.
__global__ void norms_kernel(const bf16* __restrict__ qkvm,
                             const bf16* __restrict__ qkvs,
                             float* __restrict__ qn, float* __restrict__ kn) {
  int idx = blockIdx.x * blockDim.x + threadIdx.x;
  if (idx >= NTOK * NHEAD) return;
  int h = idx % NHEAD;
  int t = idx / NHEAD;                       // t = b*L + l
  const bf16* qm = qkvm + (size_t)t * QKVN + h * DK;
  const bf16* km = qm + DMODEL;
  const bf16* qs = qkvs + (size_t)t * QKVN + h * DK;
  const bf16* ks = qs + DMODEL;
  float sq = 0.f, sk = 0.f;
#pragma unroll 8
  for (int d = 0; d < DK; ++d) {
    float a = (float)qm[d], b2 = (float)qs[d];
    float c = (float)km[d], e = (float)ks[d];
    sq += a * a + b2 * b2;
    sk += c * c + e * e;
  }
  int bb = t / LSEQ, l = t % LSEQ;
  size_t o = ((size_t)bb * NHEAD + h) * LSEQ + l;
  qn[o] = sq;
  kn[o] = sk;
}

// ---------------- generic bf16 WMMA GEMM ----------------
// C[M,N] = A[M,K] @ B[K,N] + bias; optional ReLU; optional fp32 residual add;
// outputs to fp32 (Cf) and/or bf16 (Cb). Tile: 128x64, 8 waves, K-step 32.
// A tile staged by the Tensor Data Mover (wave 0), B tile transposed by all.
__global__ __launch_bounds__(256) void gemm_bf16_kernel(
    const bf16* __restrict__ A, const bf16* __restrict__ Bw,
    const float* __restrict__ bias, const float* __restrict__ resid,
    float* __restrict__ Cf, bf16* __restrict__ Cb,
    int M, int N, int K, int relu) {
  __shared__ __align__(16) bf16 As[128][40];  // TDM-padded A slice (80B rows)
  __shared__ __align__(16) bf16 Bs[64][40];   // B^T slice: [n][k], padded

  int tid = threadIdx.x;
  int lane = tid & 31, wave = tid >> 5;
  int hi = lane >> 4, l15 = lane & 15;
  int m0 = blockIdx.y * 128;
  int n0 = blockIdx.x * 64;
  u32 as_lds = (u32)(uintptr_t)&As[0][0];

  v8f acc[4] = {v8f{}, v8f{}, v8f{}, v8f{}};

  for (int k0 = 0; k0 < K; k0 += 32) {
    __syncthreads();
    if (wave == 0) {  // async DMA of A[m0:m0+128, k0:k0+32] into padded LDS
      tdm_load_a_tile(A + (size_t)m0 * K + k0, as_lds, K);
    }
    {  // stage B transposed: 32x64 -> Bs[n][k]
      int k = tid >> 3, nb = (tid & 7) * 8;
      const bf16* src = Bw + (size_t)(k0 + k) * N + n0 + nb;
      v8bf v = *(const v8bf*)src;
      if (k0 + 32 < K) __builtin_prefetch(src + 32 * N, 0, 0);
#pragma unroll
      for (int i = 0; i < 8; ++i) Bs[nb + i][k] = v[i];
    }
    if (wave == 0) __builtin_amdgcn_s_wait_tensorcnt(0);
    __syncthreads();

    v16bf afrag = ld_frag(&As[wave * 16 + l15][0], hi);
#pragma unroll
    for (int nt = 0; nt < 4; ++nt) {
      v16bf bfrag = ld_frag(&Bs[nt * 16 + l15][0], hi);
      acc[nt] = wmma_bf16(afrag, bfrag, acc[nt]);
    }
  }

#pragma unroll
  for (int nt = 0; nt < 4; ++nt) {
    int col = n0 + nt * 16 + l15;
    float bv = bias ? bias[col] : 0.0f;
#pragma unroll
    for (int r = 0; r < 8; ++r) {
      int row = m0 + wave * 16 + hi * 8 + r;
      float v = acc[nt][r] + bv;
      if (relu) v = fmaxf(v, 0.0f);
      if (resid) v += resid[(size_t)row * N + col];
      if (Cf) Cf[(size_t)row * N + col] = v;
      if (Cb) Cb[(size_t)row * N + col] = (bf16)v;
    }
  }
}

// ---------------- streaming causal attention ----------------
// One block = 4 waves = 64 query rows of one (b,h). sim = exp(-W) in (0,1],
// so softmax = exp(sim)/sum(exp(sim)) needs no running max.
__global__ __launch_bounds__(128) void attn_kernel(
    const bf16* __restrict__ qkvm, const bf16* __restrict__ qkvs,
    const float* __restrict__ qn, const float* __restrict__ kn,
    bf16* __restrict__ am, bf16* __restrict__ as_) {
  __shared__ __align__(16) bf16 Km[32][DK];
  __shared__ __align__(16) bf16 Ks[32][DK];
  __shared__ __align__(16) bf16 Vtm[DK][40];     // [dim][key], padded
  __shared__ __align__(16) bf16 Vts[DK][40];
  __shared__ __align__(16) bf16 Psh[4][16][40];  // per-wave P tile, padded

  int tid = threadIdx.x, lane = tid & 31, wave = tid >> 5;
  int hi = lane >> 4, l15 = lane & 15;
  int b = blockIdx.z, h = blockIdx.y;
  int q0 = blockIdx.x * 64;

  // Q fragments straight from global (A layout: lane row = lane&15)
  int qrow = q0 + wave * 16 + l15;
  const bf16* qmrow = qkvm + ((size_t)b * LSEQ + qrow) * QKVN + h * DK;
  const bf16* qsrow = qkvs + ((size_t)b * LSEQ + qrow) * QKVN + h * DK;
  v16bf qmf[2], qsf[2];
#pragma unroll
  for (int ks = 0; ks < 2; ++ks) {
    qmf[ks] = ld_frag(qmrow + ks * 32, hi);
    qsf[ks] = ld_frag(qsrow + ks * 32, hi);
  }

  // per-lane rows (C layout): i = ibase + r
  int ibase = q0 + wave * 16 + hi * 8;
  size_t nb = ((size_t)b * NHEAD + h) * LSEQ;
  float qn8[8], den8[8];
#pragma unroll
  for (int r = 0; r < 8; ++r) {
    qn8[r] = qn[nb + ibase + r];
    den8[r] = 0.0f;
  }
  v8f numm[4] = {v8f{}, v8f{}, v8f{}, v8f{}};
  v8f nums[4] = {v8f{}, v8f{}, v8f{}, v8f{}};

  int jmax = q0 + 63;
  for (int j0 = 0; j0 <= jmax; j0 += 32) {
    __syncthreads();
    {  // stage K tiles (row-major) and V tiles (transposed)
      int key = tid >> 2, dk0 = (tid & 3) * 16;
      size_t kb = ((size_t)b * LSEQ + j0 + key) * QKVN + DMODEL + h * DK + dk0;
      *(v8bf*)&Km[key][dk0]     = *(const v8bf*)(qkvm + kb);
      *(v8bf*)&Km[key][dk0 + 8] = *(const v8bf*)(qkvm + kb + 8);
      *(v8bf*)&Ks[key][dk0]     = *(const v8bf*)(qkvs + kb);
      *(v8bf*)&Ks[key][dk0 + 8] = *(const v8bf*)(qkvs + kb + 8);
      size_t vb = ((size_t)b * LSEQ + j0 + key) * QKVN + 2 * DMODEL + h * DK + dk0;
      v8bf vm0 = *(const v8bf*)(qkvm + vb);
      v8bf vm1 = *(const v8bf*)(qkvm + vb + 8);
      v8bf vs0 = *(const v8bf*)(qkvs + vb);
      v8bf vs1 = *(const v8bf*)(qkvs + vb + 8);
#pragma unroll
      for (int i = 0; i < 8; ++i) {
        Vtm[dk0 + i][key] = vm0[i];
        Vtm[dk0 + 8 + i][key] = vm1[i];
        Vts[dk0 + i][key] = vs0[i];
        Vts[dk0 + 8 + i][key] = vs1[i];
      }
    }
    __syncthreads();

    // S = Qm Km^T + Qs Ks^T   (two 16x16 key subtiles, dk in 2 K-steps)
    v8f sacc[2] = {v8f{}, v8f{}};
#pragma unroll
    for (int nt = 0; nt < 2; ++nt) {
#pragma unroll
      for (int ks = 0; ks < 2; ++ks) {
        v16bf bm = ld_frag(&Km[nt * 16 + l15][ks * 32], hi);
        sacc[nt] = wmma_bf16(qmf[ks], bm, sacc[nt]);
        v16bf bs = ld_frag(&Ks[nt * 16 + l15][ks * 32], hi);
        sacc[nt] = wmma_bf16(qsf[ks], bs, sacc[nt]);
      }
    }

    // P = causal ? exp(exp(-sqrt(max(qn+kn-2S,0)))) : 0
#pragma unroll
    for (int nt = 0; nt < 2; ++nt) {
      int j = j0 + nt * 16 + l15;
      float knj = kn[nb + j];
#pragma unroll
      for (int r = 0; r < 8; ++r) {
        int i = ibase + r;
        float w2 = fmaxf(qn8[r] + knj - 2.0f * sacc[nt][r], 0.0f);
        float sim = __expf(-sqrtf(w2));
        float e = (j <= i) ? __expf(sim) : 0.0f;
        den8[r] += e;
        Psh[wave][hi * 8 + r][nt * 16 + l15] = (bf16)e;
      }
    }

    // num += P @ V (both value streams share the P tile)
    v16bf pf = ld_frag(&Psh[wave][l15][0], hi);
#pragma unroll
    for (int nt2 = 0; nt2 < 4; ++nt2) {
      v16bf bm = ld_frag(&Vtm[nt2 * 16 + l15][0], hi);
      numm[nt2] = wmma_bf16(pf, bm, numm[nt2]);
      v16bf bs = ld_frag(&Vts[nt2 * 16 + l15][0], hi);
      nums[nt2] = wmma_bf16(pf, bs, nums[nt2]);
    }
  }

  // reduce denominators across the 16 lanes sharing each row
#pragma unroll
  for (int r = 0; r < 8; ++r) {
    float s = den8[r];
    s += __shfl_xor(s, 1, 32);
    s += __shfl_xor(s, 2, 32);
    s += __shfl_xor(s, 4, 32);
    s += __shfl_xor(s, 8, 32);
    den8[r] = s;
  }

  // write merged-head attn outputs as bf16
#pragma unroll
  for (int nt2 = 0; nt2 < 4; ++nt2) {
    int dim = h * DK + nt2 * 16 + l15;
#pragma unroll
    for (int r = 0; r < 8; ++r) {
      int l = ibase + r;
      size_t o = ((size_t)b * LSEQ + l) * DMODEL + dim;
      float inv = 1.0f / den8[r];
      am[o]  = (bf16)(numm[nt2][r] * inv);
      as_[o] = (bf16)(nums[nt2][r] * inv);
    }
  }
}

// ---------------- host-side orchestration ----------------

static inline void cvt(const float* x, bf16* y, int n, hipStream_t s) {
  cvt_bf16_kernel<<<(n + 255) / 256, 256, 0, s>>>(x, y, n);
}

extern "C" void kernel_launch(void* const* d_in, const int* in_sizes, int n_in,
                              void* d_out, int out_size, void* d_ws, size_t ws_size,
                              hipStream_t stream) {
  const float* mu      = (const float*)d_in[0];
  const float* sigma   = (const float*)d_in[1];
  const float* w_qkv_mu = (const float*)d_in[2];
  const float* b_qkv_mu = (const float*)d_in[3];
  const float* w_qkv_sg = (const float*)d_in[4];
  const float* b_qkv_sg = (const float*)d_in[5];
  const float* w_out   = (const float*)d_in[6];
  const float* b_out   = (const float*)d_in[7];
  const float* ln1_g   = (const float*)d_in[8];
  const float* ln1_b   = (const float*)d_in[9];
  const float* ln2_g   = (const float*)d_in[10];
  const float* ln2_b   = (const float*)d_in[11];
  const float* ffm_w1  = (const float*)d_in[12];
  const float* ffm_b1  = (const float*)d_in[13];
  const float* ffm_w2  = (const float*)d_in[14];
  const float* ffm_b2  = (const float*)d_in[15];
  const float* ffs_w1  = (const float*)d_in[16];
  const float* ffs_b1  = (const float*)d_in[17];
  const float* ffs_w2  = (const float*)d_in[18];
  const float* ffs_b2  = (const float*)d_in[19];
  float* out = (float*)d_out;

  // carve workspace
  char* p = (char*)d_ws;
  auto carve = [&](size_t bytes) -> void* {
    void* r = (void*)p;
    p += (bytes + 255) & ~(size_t)255;
    return r;
  };
  bf16* wqkvm_b = (bf16*)carve((size_t)DMODEL * QKVN * 2);
  bf16* wqkvs_b = (bf16*)carve((size_t)DMODEL * QKVN * 2);
  bf16* wout_b  = (bf16*)carve((size_t)DMODEL * DMODEL * 2);
  bf16* wm1_b   = (bf16*)carve((size_t)DMODEL * DFF * 2);
  bf16* wm2_b   = (bf16*)carve((size_t)DFF * DMODEL * 2);
  bf16* ws1_b   = (bf16*)carve((size_t)DMODEL * DFF * 2);
  bf16* ws2_b   = (bf16*)carve((size_t)DFF * DMODEL * 2);
  bf16* xm      = (bf16*)carve((size_t)NTOK * DMODEL * 2);
  bf16* xs      = (bf16*)carve((size_t)NTOK * DMODEL * 2);
  bf16* qkvm    = (bf16*)carve((size_t)NTOK * QKVN * 2);
  bf16* qkvs    = (bf16*)carve((size_t)NTOK * QKVN * 2);
  float* qn     = (float*)carve((size_t)NTOK * NHEAD * 4);
  float* kn     = (float*)carve((size_t)NTOK * NHEAD * 4);
  bf16* am      = (bf16*)carve((size_t)NTOK * DMODEL * 2);
  bf16* as_     = (bf16*)carve((size_t)NTOK * DMODEL * 2);
  float* mu2    = (float*)carve((size_t)NTOK * DMODEL * 4);
  float* sg2    = (float*)carve((size_t)NTOK * DMODEL * 4);
  bf16* hm      = (bf16*)carve((size_t)NTOK * DMODEL * 2);
  bf16* hs      = (bf16*)carve((size_t)NTOK * DMODEL * 2);
  bf16* tbuf    = (bf16*)carve((size_t)NTOK * DFF * 2);

  // 1) weights -> bf16
  cvt(w_qkv_mu, wqkvm_b, DMODEL * QKVN, stream);
  cvt(w_qkv_sg, wqkvs_b, DMODEL * QKVN, stream);
  cvt(w_out,    wout_b,  DMODEL * DMODEL, stream);
  cvt(ffm_w1,   wm1_b,   DMODEL * DFF, stream);
  cvt(ffm_w2,   wm2_b,   DFF * DMODEL, stream);
  cvt(ffs_w1,   ws1_b,   DMODEL * DFF, stream);
  cvt(ffs_w2,   ws2_b,   DFF * DMODEL, stream);

  // 2) LN(mu) and softplus(sigma) -> bf16
  ln_bf16_kernel<<<NTOK, 256, 0, stream>>>(mu, ln1_g, ln1_b, xm);
  softplus_bf16_kernel<<<(NTOK * DMODEL + 255) / 256, 256, 0, stream>>>(sigma, xs, NTOK * DMODEL);

  // 3) QKV projections (bf16 out)
  gemm_bf16_kernel<<<dim3(QKVN / 64, NTOK / 128), 256, 0, stream>>>(
      xm, wqkvm_b, b_qkv_mu, nullptr, nullptr, qkvm, NTOK, QKVN, DMODEL, 0);
  gemm_bf16_kernel<<<dim3(QKVN / 64, NTOK / 128), 256, 0, stream>>>(
      xs, wqkvs_b, b_qkv_sg, nullptr, nullptr, qkvs, NTOK, QKVN, DMODEL, 0);

  // 4) per-(token,head) squared norms
  norms_kernel<<<(NTOK * NHEAD + 255) / 256, 256, 0, stream>>>(qkvm, qkvs, qn, kn);

  // 5) attention
  attn_kernel<<<dim3(LSEQ / 64, NHEAD, BSZ), 128, 0, stream>>>(qkvm, qkvs, qn, kn, am, as_);

  // 6) out projection + residual -> mu2/sg2 (fp32)
  gemm_bf16_kernel<<<dim3(DMODEL / 64, NTOK / 128), 256, 0, stream>>>(
      am, wout_b, b_out, mu, mu2, nullptr, NTOK, DMODEL, DMODEL, 0);
  gemm_bf16_kernel<<<dim3(DMODEL / 64, NTOK / 128), 256, 0, stream>>>(
      as_, wout_b, b_out, sigma, sg2, nullptr, NTOK, DMODEL, DMODEL, 0);

  // 7) LN2 / softplus -> bf16 FFN inputs
  ln_bf16_kernel<<<NTOK, 256, 0, stream>>>(mu2, ln2_g, ln2_b, hm);
  softplus_bf16_kernel<<<(NTOK * DMODEL + 255) / 256, 256, 0, stream>>>(sg2, hs, NTOK * DMODEL);

  // 8) mu FFN: relu(hm@w1+b1) -> tbuf; tbuf@w2+b2+mu2 -> out[0]
  gemm_bf16_kernel<<<dim3(DFF / 64, NTOK / 128), 256, 0, stream>>>(
      hm, wm1_b, ffm_b1, nullptr, nullptr, tbuf, NTOK, DFF, DMODEL, 1);
  gemm_bf16_kernel<<<dim3(DMODEL / 64, NTOK / 128), 256, 0, stream>>>(
      tbuf, wm2_b, ffm_b2, mu2, out, nullptr, NTOK, DMODEL, DFF, 0);

  // 9) sigma FFN: relu(hs@w1+b1) -> tbuf; tbuf@w2+b2+sg2 -> out[1]
  gemm_bf16_kernel<<<dim3(DFF / 64, NTOK / 128), 256, 0, stream>>>(
      hs, ws1_b, ffs_b1, nullptr, nullptr, tbuf, NTOK, DFF, DMODEL, 1);
  gemm_bf16_kernel<<<dim3(DMODEL / 64, NTOK / 128), 256, 0, stream>>>(
      tbuf, ws2_b, ffs_b2, sg2, out + (size_t)NTOK * DMODEL, nullptr, NTOK, DMODEL, DFF, 0);
}
